// DynamicChunkingLayer_2293512536844
// MI455X (gfx1250) — matrix operand order repaired and checked by reading, not compiled, not run on hardware
//
#include <hip/hip_runtime.h>
#include <hip/hip_bf16.h>
#include <math.h>

// ---------------- problem constants (from reference) ----------------
static constexpr int Bn = 16;
static constexpr int Sn = 8192;
static constexpr int Dn = 256;
static constexpr float EPSF = 1e-8f;
static constexpr float TARGET = (float)(Sn / 4);   // CHUNK_RATIO = 4

// ---------------- CDNA5 WMMA types ----------------
typedef __attribute__((ext_vector_type(16))) __bf16 v16bf;
typedef __attribute__((ext_vector_type(8)))  __bf16 v8bf;
typedef __attribute__((ext_vector_type(4)))  __bf16 v4bf;
typedef __attribute__((ext_vector_type(8)))  float  v8f;

__device__ __forceinline__ v8f wmma_bf16(v16bf a, v16bf b, v8f c) {
    // D = A(16x32 bf16) x B(32x16 bf16) + C(16x16 f32)
    return __builtin_amdgcn_wmma_f32_16x16x32_bf16(
        /*neg_a=*/false, a, /*neg_b=*/false, b,
        /*c_mod=*/(short)0, c, /*reuse_a=*/false, /*reuse_b=*/false);
}

// A fragment (16x32 bf16, MxK):
//   lanes 0-15 : M=lane,   K in {0..15}
//   lanes 16-31: M=lane-16, K in {16..31}
// row_base points at LDS row M (row-major, Dn bf16 per row).
__device__ __forceinline__ v16bf load_A(const __bf16* row_base, int ks, int khi) {
    return *(const v16bf*)(row_base + ks * 32 + 2 * khi);
}

// B fragment (32x16 bf16, KxN): lane holds column N = lane&15,
// K = (lane>=16 ? 16 : 0) + e, e=0..15. Weights stored N-major (transposed),
// so the 16 K-values are 32 contiguous bytes.
__device__ __forceinline__ v16bf load_B(const __bf16* Wt, int ntile, int lane, int ks) {
    const int n = ntile * 16 + (lane & 15);
    const int kbase = ks * 32 + ((lane >> 4) << 4);
    return *(const v16bf*)(Wt + (size_t)n * Dn + kbase);
}

__device__ __forceinline__ float sigmoidf(float v) { return 1.f / (1.f + expf(-v)); }

// ---------------- kernel 1: weight convert + transpose (f32 -> bf16, N-major) ----
__global__ __launch_bounds__(256)
void wconv_kernel(const float* __restrict__ Wq, const float* __restrict__ Wk,
                  __bf16* __restrict__ WqT, __bf16* __restrict__ WkT) {
    const int n = blockIdx.x;     // 0..255 output column
    const int k = threadIdx.x;    // 0..255 reduction index
    WqT[(size_t)n * Dn + k] = (__bf16)Wq[(size_t)k * Dn + n];
    WkT[(size_t)n * Dn + k] = (__bf16)Wk[(size_t)k * Dn + n];
}

// ---------------- kernel 2: fused GEMM (WMMA bf16) + cosine + vol/volu --------
// One workgroup (256 thr = 8 waves) per 16-row M-tile of the flattened (B*S) rows.
// q-tile rows [r0, r0+16); k-tile rows [r0-1, r0+15)  (one-row shift => dot is
// a same-row elementwise product of the two C fragments).
__global__ __launch_bounds__(256)
void boundary_kernel(const float* __restrict__ x,
                     const __bf16* __restrict__ WqT, const __bf16* __restrict__ WkT,
                     const float* __restrict__ bq, const float* __restrict__ bk,
                     const float* __restrict__ w_vol, const float* __restrict__ b_vol,
                     const float* __restrict__ w_volu, const float* __restrict__ b_volu,
                     float* __restrict__ probs_out, int* __restrict__ hard_out) {
    __shared__ __bf16 xq_lds[16 * Dn];   // A fragments source, q rows
    __shared__ __bf16 xk_lds[16 * Dn];   // A fragments source, k rows (shifted -1)
    __shared__ float s_dot[16], s_q2[16], s_k2[16], s_vol[16], s_volu[16];

    const int t = threadIdx.x;
    const int r0 = blockIdx.x * 16;

    if (t < 16) { s_dot[t] = 0.f; s_q2[t] = 0.f; s_k2[t] = 0.f; s_vol[t] = 0.f; s_volu[t] = 0.f; }
    __syncthreads();

    // ---- stage x -> LDS (bf16), fuse vol/volu dot partials on the q rows ----
    {
        const int m  = t >> 4;          // row in tile, 0..15
        const int c0 = (t & 15) * 16;   // 16 contiguous columns per thread

        const float4* xr = (const float4*)(x + (size_t)(r0 + m) * Dn);
        const float4* wv = (const float4*)w_vol;
        const float4* wu = (const float4*)w_volu;
        float pv = 0.f, pu = 0.f;
#pragma unroll
        for (int g = 0; g < 4; ++g) {
            const float4 v = xr[(c0 >> 2) + g];
            const float4 a = wv[(c0 >> 2) + g];
            const float4 b = wu[(c0 >> 2) + g];
            pv += v.x * a.x + v.y * a.y + v.z * a.z + v.w * a.w;
            pu += v.x * b.x + v.y * b.y + v.z * b.z + v.w * b.w;
            v4bf p; p[0] = (__bf16)v.x; p[1] = (__bf16)v.y; p[2] = (__bf16)v.z; p[3] = (__bf16)v.w;
            *(v4bf*)&xq_lds[m * Dn + c0 + g * 4] = p;
        }
        atomicAdd(&s_vol[m], pv);
        atomicAdd(&s_volu[m], pu);

        int rk = r0 - 1 + m; if (rk < 0) rk = 0;   // clamp (row masked below)
        const float4* xr2 = (const float4*)(x + (size_t)rk * Dn);
#pragma unroll
        for (int g = 0; g < 4; ++g) {
            const float4 v = xr2[(c0 >> 2) + g];
            v4bf p; p[0] = (__bf16)v.x; p[1] = (__bf16)v.y; p[2] = (__bf16)v.z; p[3] = (__bf16)v.w;
            *(v4bf*)&xk_lds[m * Dn + c0 + g * 4] = p;
        }
        // prefetch next M-tile's x rows toward L2
        if (r0 + 16 + m < Bn * Sn)
            __builtin_prefetch(x + (size_t)(r0 + 16 + m) * Dn + c0, 0, 1);
    }
    __syncthreads();

    // ---- WMMA GEMMs: each wave owns N-tiles {w, w+8} for BOTH q and k ----
    const int wave = t >> 5;
    const int lane = t & 31;
    const int n0 = wave, n1 = wave + 8;
    const int khi = (lane >> 4) << 3;                  // 0 or 8
    const __bf16* aq_base = &xq_lds[(lane & 15) * Dn]; // A row for this lane
    const __bf16* ak_base = &xk_lds[(lane & 15) * Dn];

    v8f q0 = {}, q1 = {}, k0 = {}, k1 = {};
#pragma unroll
    for (int ks = 0; ks < Dn / 32; ++ks) {
        const v16bf aq = load_A(aq_base, ks, khi);
        const v16bf ak = load_A(ak_base, ks, khi);
        q0 = wmma_bf16(aq, load_B(WqT, n0, lane, ks), q0);
        q1 = wmma_bf16(aq, load_B(WqT, n1, lane, ks), q1);
        k0 = wmma_bf16(ak, load_B(WkT, n0, lane, ks), k0);
        k1 = wmma_bf16(ak, load_B(WkT, n1, lane, ks), k1);
    }

    // bias add (C/D layout: lane&15 = N, VGPR v = row M, +8 for upper half-wave)
    const float bq0 = bq[n0 * 16 + (lane & 15)], bq1 = bq[n1 * 16 + (lane & 15)];
    const float bk0 = bk[n0 * 16 + (lane & 15)], bk1 = bk[n1 * 16 + (lane & 15)];
#pragma unroll
    for (int v = 0; v < 8; ++v) { q0[v] += bq0; q1[v] += bq1; k0[v] += bk0; k1[v] += bk1; }

    // row-wise reductions across N inside the fragment layout
#pragma unroll
    for (int v = 0; v < 8; ++v) {
        float d  = q0[v] * k0[v] + q1[v] * k1[v];
        float qs = q0[v] * q0[v] + q1[v] * q1[v];
        float ks_ = k0[v] * k0[v] + k1[v] * k1[v];
#pragma unroll
        for (int off = 1; off < 16; off <<= 1) {
            d   += __shfl_xor(d,   off, 32);
            qs  += __shfl_xor(qs,  off, 32);
            ks_ += __shfl_xor(ks_, off, 32);
        }
        if ((lane & 15) == 0) {
            const int m = v + ((lane >> 4) << 3);
            atomicAdd(&s_dot[m], d);     // ds_add_f32
            atomicAdd(&s_q2[m],  qs);
            atomicAdd(&s_k2[m],  ks_);
        }
    }
    __syncthreads();

    // ---- finalize 16 rows ----
    if (t < 16) {
        const int sg = r0 + t;            // flat row in (B*S)
        const int sb = sg & (Sn - 1);     // position within batch
        float combined;
        if (sb == 0) {
            combined = 1.f;               // concat ones column
        } else {
            float nq = sqrtf(s_q2[t]); nq = fmaxf(nq, EPSF);
            float nk = sqrtf(s_k2[t]); nk = fmaxf(nk, EPSF);
            const float sim = s_dot[t] / (nq * nk);
            const float vol  = sigmoidf(s_vol[t]  + b_vol[0]);
            const float volu = sigmoidf(s_volu[t] + b_volu[0]);
            combined = (1.f - sim) + 0.3f * vol + 0.2f * volu;
        }
        const float p = sigmoidf(combined);
        probs_out[sg] = p;
        hard_out[sg] = (p > 0.5f) ? 1 : 0;
    }
}

// ---------------- kernel 3: per-batch scan of hard -> compacted positions ----
__global__ __launch_bounds__(256)
void scan_kernel(const int* __restrict__ hard, int* __restrict__ pos,
                 int* __restrict__ counts) {
    const int b = blockIdx.x, t = threadIdx.x;
    const int* h = hard + (size_t)b * Sn;
    const int per = Sn / 256;             // 32 elements per thread

    int psum = 0;
    for (int i = 0; i < per; ++i) psum += h[t * per + i];

    __shared__ int ss[256];
    ss[t] = psum;
    __syncthreads();
    for (int off = 1; off < 256; off <<= 1) {
        int v = (t >= off) ? ss[t - off] : 0;
        __syncthreads();
        ss[t] += v;
        __syncthreads();
    }
    int run = ss[t] - psum;               // exclusive prefix
    for (int i = 0; i < per; ++i) {
        const int s = t * per + i;
        if (h[s]) { pos[(size_t)b * Sn + run] = s; ++run; }
    }
    if (t == 255) counts[b] = ss[255];
}

// ---------------- kernel 4: ratio loss ----------------
__global__ void loss_kernel(const int* __restrict__ counts, float* __restrict__ out_loss) {
    if (threadIdx.x == 0 && blockIdx.x == 0) {
        float sum = 0.f;
        for (int b = 0; b < Bn; ++b) sum += (float)counts[b];
        const float d = sum / (float)Bn - TARGET;
        *out_loss = d * d;
    }
}

// ---------------- kernel 5: single-pass compressed output ----------------
// Each output row written exactly once: selected x row if j < count, else zeros.
__global__ __launch_bounds__(256)
void gather_kernel(const float* __restrict__ x, const int* __restrict__ pos,
                   const int* __restrict__ counts, float* __restrict__ out) {
    const int row = blockIdx.x * 4 + (threadIdx.x >> 6);  // 4 rows / block
    const int l = threadIdx.x & 63;                       // 64 lanes x float4 = 256
    const int b = row >> 13;                              // row / Sn
    const int j = row & (Sn - 1);
    float4 v = make_float4(0.f, 0.f, 0.f, 0.f);
    if (j < counts[b]) {
        const int src = b * Sn + pos[(size_t)b * Sn + j];
        v = ((const float4*)x)[(size_t)src * (Dn / 4) + l];
    }
    ((float4*)out)[(size_t)row * (Dn / 4) + l] = v;
}

// ---------------- host launcher ----------------
extern "C" void kernel_launch(void* const* d_in, const int* in_sizes, int n_in,
                              void* d_out, int out_size, void* d_ws, size_t ws_size,
                              hipStream_t stream) {
    const float* x      = (const float*)d_in[0];
    const float* Wq     = (const float*)d_in[1];
    const float* bq     = (const float*)d_in[2];
    const float* Wk     = (const float*)d_in[3];
    const float* bk     = (const float*)d_in[4];
    const float* w_vol  = (const float*)d_in[5];
    const float* b_vol  = (const float*)d_in[6];
    const float* w_volu = (const float*)d_in[7];
    const float* b_volu = (const float*)d_in[8];

    float* out        = (float*)d_out;
    float* compressed = out;                                  // (B,S,D)
    float* probs      = out + (size_t)Bn * Sn * Dn;           // (B,S)
    float* loss       = probs + (size_t)Bn * Sn;              // scalar

    char* ws = (char*)d_ws;
    __bf16* WqT  = (__bf16*)ws;                               // D*D bf16
    __bf16* WkT  = (__bf16*)(ws + (size_t)Dn * Dn * 2);       // D*D bf16
    int* hard    = (int*)(ws + 2 * (size_t)Dn * Dn * 2);      // B*S int
    int* pos     = hard + (size_t)Bn * Sn;                    // B*S int
    int* counts  = pos + (size_t)Bn * Sn;                     // B int

    wconv_kernel<<<Dn, Dn, 0, stream>>>(Wq, Wk, WqT, WkT);
    boundary_kernel<<<(Bn * Sn) / 16, 256, 0, stream>>>(
        x, WqT, WkT, bq, bk, w_vol, b_vol, w_volu, b_volu, probs, hard);
    scan_kernel<<<Bn, 256, 0, stream>>>(hard, pos, counts);
    loss_kernel<<<1, 32, 0, stream>>>(counts, loss);
    gather_kernel<<<(Bn * Sn) / 4, 256, 0, stream>>>(x, pos, counts, compressed);
}
